// CoreMLLinear2Bit_51307679318807
// MI455X (gfx1250) — matrix operand order repaired and checked by reading, not compiled, not run on hardware
//
#include <hip/hip_runtime.h>

typedef _Float16     v16h  __attribute__((ext_vector_type(16)));
typedef float        v8f   __attribute__((ext_vector_type(8)));
typedef unsigned int u32x8 __attribute__((ext_vector_type(8)));
typedef unsigned int u32x4 __attribute__((ext_vector_type(4)));

#define IN_F    4096
#define OUT_F   12288
#define M_TOTAL 4096
#define BM      128
#define BN      128
#define BK      32
#define KT      (IN_F / BK)   // 128 K-steps
#define BKU     (BK / 2)      // 16 dwords per f16 LDS row

__device__ __forceinline__ unsigned int pack2(float a, float b) {
    union { _Float16 h[2]; unsigned int u; } q;
    q.h[0] = (_Float16)a; q.h[1] = (_Float16)b;
    return q.u;
}

// ISA 16-bit A layout: VGPR v<4 -> k = hi*8 + 2v ; v>=4 -> k = 16 + hi*8 + 2(v-4)
__device__ __forceinline__ v16h frag_a16(const unsigned int* base, int row, int hi) {
    const u32x4 a = *(const u32x4*)(base + row * BKU + hi * 4);
    const u32x4 b = *(const u32x4*)(base + row * BKU + 8 + hi * 4);
    u32x8 u;
    u[0] = a[0]; u[1] = a[1]; u[2] = a[2]; u[3] = a[3];
    u[4] = b[0]; u[5] = b[1]; u[6] = b[2]; u[7] = b[3];
    return __builtin_bit_cast(v16h, u);
}
// ISA 16-bit B layout: VGPR v<4 -> k = hi*8 + 2v ; v>=4 -> k = 16 + hi*8 + 2(v-4)
__device__ __forceinline__ v16h frag_b16(const unsigned int* base, int row, int hi) {
    const u32x4 a = *(const u32x4*)(base + row * BKU + hi * 4);
    const u32x4 b = *(const u32x4*)(base + row * BKU + 8 + hi * 4);
    u32x8 u;
    u[0] = a[0]; u[1] = a[1]; u[2] = a[2]; u[3] = a[3];
    u[4] = b[0]; u[5] = b[1]; u[6] = b[2]; u[7] = b[3];
    return __builtin_bit_cast(v16h, u);
}

// ---------------------------------------------------------------------------
// Prep kernel 1: x f32 -> f16 (row-major, WMMA-ready packing)
// ---------------------------------------------------------------------------
__global__ __launch_bounds__(256)
void cvt_x_kernel(const float* __restrict__ x, _Float16* __restrict__ xh) {
    const size_t i = ((size_t)blockIdx.x * 256 + threadIdx.x) * 8;
    const float4 f0 = *(const float4*)(x + i);
    const float4 f1 = *(const float4*)(x + i + 4);
    u32x4 p;
    p[0] = pack2(f0.x, f0.y); p[1] = pack2(f0.z, f0.w);
    p[2] = pack2(f1.x, f1.y); p[3] = pack2(f1.z, f1.w);
    *(u32x4*)(xh + i) = p;
}

// ---------------------------------------------------------------------------
// Prep kernel 2: dequantize W once: wh[o][k] = f16(levels[idx[o][k]] * scale[o])
// One block per output row; perm-LUT dequant, 2 weights per v_perm_b32.
// ---------------------------------------------------------------------------
__global__ __launch_bounds__(256)
void deq_w_kernel(const int*   __restrict__ widx,
                  const float* __restrict__ levels,
                  const float* __restrict__ wscale,
                  _Float16*    __restrict__ wh) {
    const unsigned row = blockIdx.x;            // 0..OUT_F-1
    const unsigned k0  = threadIdx.x * 16;      // 16 weights per thread
    const float scl = wscale[row];
    const unsigned htab_lo = pack2(levels[0] * scl, levels[1] * scl);
    const unsigned htab_hi = pack2(levels[2] * scl, levels[3] * scl);
    auto deq_pair = [&](int va, int vb) -> unsigned int {
        const unsigned sel = 0x01000100u + (unsigned)va * 0x202u
                                         + (unsigned)vb * 0x02020000u;
        return __builtin_amdgcn_perm(htab_hi, htab_lo, sel);
    };
    const int4* bp = (const int4*)(widx + (size_t)row * IN_F + k0);
    const int4 b0 = bp[0], b1 = bp[1], b2 = bp[2], b3 = bp[3];
    u32x4 p0, p1;
    p0[0] = deq_pair(b0.x, b0.y); p0[1] = deq_pair(b0.z, b0.w);
    p0[2] = deq_pair(b1.x, b1.y); p0[3] = deq_pair(b1.z, b1.w);
    p1[0] = deq_pair(b2.x, b2.y); p1[1] = deq_pair(b2.z, b2.w);
    p1[2] = deq_pair(b3.x, b3.y); p1[3] = deq_pair(b3.z, b3.w);
    _Float16* dst = wh + (size_t)row * IN_F + k0;
    *(u32x4*)(dst)     = p0;
    *(u32x4*)(dst + 8) = p1;
}

// ---------------------------------------------------------------------------
// Main GEMM: both tiles DMA'd by global_load_async_to_lds_b128, pure WMMA loop
// ---------------------------------------------------------------------------
__global__ __launch_bounds__(256)
void q2_gemm_async(const _Float16* __restrict__ xh,
                   const _Float16* __restrict__ wh,
                   const float*    __restrict__ bias,
                   float*          __restrict__ out)
{
    __shared__ unsigned int Ash[2][BM * BKU];   // 8 KB each buffer
    __shared__ unsigned int Bsh[2][BN * BKU];   // 8 KB each buffer

    const int tid  = threadIdx.x;
    const int lane = tid & 31;
    const int wave = tid >> 5;
    const int wm   = wave & 3;        // M sub-block: wm*32
    const int wn   = wave >> 2;       // N sub-block: wn*64
    const int l15  = lane & 15;
    const int hi   = lane >> 4;

    const int bm = blockIdx.x * BM;   // M fastest -> idx/W streams HBM once
    const int bn = blockIdx.y * BN;

    const int lrow = tid >> 1;            // 0..127
    const int lseg = (tid & 1) * 16;      // k offset 0 or 16

    // per-lane LDS byte addresses of this thread's 32-byte slot, both buffers
    const unsigned a_lds[2] = {
        (unsigned)(size_t)&Ash[0][lrow * BKU + (lseg >> 1)],
        (unsigned)(size_t)&Ash[1][lrow * BKU + (lseg >> 1)] };
    const unsigned b_lds[2] = {
        (unsigned)(size_t)&Bsh[0][lrow * BKU + (lseg >> 1)],
        (unsigned)(size_t)&Bsh[1][lrow * BKU + (lseg >> 1)] };
    // per-lane global byte offsets (advance 64 B per K-step)
    const unsigned a_off0 = ((unsigned)(bm + lrow) * IN_F + (unsigned)lseg) * 2u;
    const unsigned b_off0 = ((unsigned)(bn + lrow) * IN_F + (unsigned)lseg) * 2u;

    // DMA one A-tile + one B-tile into LDS buffer `buf` (ASYNCcnt-tracked);
    // imm offset advances BOTH the LDS and global addresses (ISA 15.18.3).
    auto fetch = [&](int buf, int kt) {
        const unsigned ka = a_off0 + (unsigned)kt * (BK * 2);
        const unsigned kb = b_off0 + (unsigned)kt * (BK * 2);
        asm volatile(
            "global_load_async_to_lds_b128 %0, %2, %4 offset:0\n\t"
            "global_load_async_to_lds_b128 %0, %2, %4 offset:16\n\t"
            "global_load_async_to_lds_b128 %1, %3, %5 offset:0\n\t"
            "global_load_async_to_lds_b128 %1, %3, %5 offset:16"
            :: "v"(a_lds[buf]), "v"(b_lds[buf]), "v"(ka), "v"(kb),
               "s"(xh), "s"(wh)
            : "memory");
    };

    v8f acc[2][4];
#pragma unroll
    for (int i = 0; i < 2; ++i)
#pragma unroll
        for (int j = 0; j < 4; ++j) acc[i][j] = v8f{};

    // prologue
    fetch(0, 0);
    asm volatile("s_wait_asynccnt 0x0" ::: "memory");
    __syncthreads();

    for (int kt = 0; kt < KT; ++kt) {
        const int cur = kt & 1;
        if (kt + 1 < KT) fetch(cur ^ 1, kt + 1);       // DMA next tiles

        v16h af[2], bf[4];
#pragma unroll
        for (int mi = 0; mi < 2; ++mi)
            af[mi] = frag_a16(Ash[cur], wm * 32 + mi * 16 + l15, hi);
#pragma unroll
        for (int ni = 0; ni < 4; ++ni)
            bf[ni] = frag_b16(Bsh[cur], wn * 64 + ni * 16 + l15, hi);

#pragma unroll
        for (int mi = 0; mi < 2; ++mi)
#pragma unroll
            for (int ni = 0; ni < 4; ++ni)
                acc[mi][ni] = __builtin_amdgcn_wmma_f32_16x16x32_f16(
                    false, af[mi], false, bf[ni],
                    (short)0, acc[mi][ni], false, false);

        asm volatile("s_wait_asynccnt 0x0" ::: "memory");
        __syncthreads();
    }

    // epilogue: D layout -> row = mi*16 + hi*8 + r, col = ni*16 + l15
#pragma unroll
    for (int mi = 0; mi < 2; ++mi) {
        const int row0 = bm + wm * 32 + mi * 16 + hi * 8;
#pragma unroll
        for (int ni = 0; ni < 4; ++ni) {
            const int col = bn + wn * 64 + ni * 16 + l15;
            const float bv = bias[col];
            float* op = out + (size_t)row0 * OUT_F + col;
#pragma unroll
            for (int r = 0; r < 8; ++r)
                op[(size_t)r * OUT_F] = acc[mi][ni][r] + bv;
        }
    }
}

// ---------------------------------------------------------------------------
// Fallback (ws too small): fused kernel, register-staged loads, perm dequant
// ---------------------------------------------------------------------------
__global__ __launch_bounds__(256)
void q2_linear_fused(const float* __restrict__ x,
                     const float* __restrict__ levels,
                     const int*   __restrict__ widx,
                     const float* __restrict__ wscale,
                     const float* __restrict__ bias,
                     float*       __restrict__ out)
{
    __shared__ unsigned int Ash[2][BM * BKU];
    __shared__ unsigned int Bsh[2][BN * BKU];

    const int tid  = threadIdx.x;
    const int lane = tid & 31;
    const int wave = tid >> 5;
    const int wm   = wave & 3;
    const int wn   = wave >> 2;
    const int l15  = lane & 15;
    const int hi   = lane >> 4;

    const int bm = blockIdx.x * BM;
    const int bn = blockIdx.y * BN;

    const int lrow = tid >> 1;
    const int lseg = (tid & 1) * 16;

    const float scl = wscale[bn + lrow];
    const unsigned htab_lo = pack2(levels[0] * scl, levels[1] * scl);
    const unsigned htab_hi = pack2(levels[2] * scl, levels[3] * scl);
    auto deq_pair = [&](int va, int vb) -> unsigned int {
        const unsigned sel = 0x01000100u + (unsigned)va * 0x202u
                                         + (unsigned)vb * 0x02020000u;
        return __builtin_amdgcn_perm(htab_hi, htab_lo, sel);
    };

    const float* xrow = x    + (size_t)(bm + lrow) * IN_F + lseg;
    const int*   irow = widx + (size_t)(bn + lrow) * IN_F + lseg;

    float4 ar[4]; int4 br[4];
    v8f acc[2][4];
#pragma unroll
    for (int i = 0; i < 2; ++i)
#pragma unroll
        for (int j = 0; j < 4; ++j) acc[i][j] = v8f{};

    auto load_regs = [&](int kt) {
        const float4* ap = (const float4*)(xrow + kt * BK);
        const int4*   bp = (const int4*)(irow + kt * BK);
#pragma unroll
        for (int i = 0; i < 4; ++i) { ar[i] = ap[i]; br[i] = bp[i]; }
    };
    auto store_lds = [&](int buf) {
        unsigned int* adst = &Ash[buf][lrow * BKU + (lseg >> 1)];
        unsigned int* bdst = &Bsh[buf][lrow * BKU + (lseg >> 1)];
        u32x4 pa0, pa1, pb0, pb1;
        pa0[0] = pack2(ar[0].x, ar[0].y); pa0[1] = pack2(ar[0].z, ar[0].w);
        pa0[2] = pack2(ar[1].x, ar[1].y); pa0[3] = pack2(ar[1].z, ar[1].w);
        pa1[0] = pack2(ar[2].x, ar[2].y); pa1[1] = pack2(ar[2].z, ar[2].w);
        pa1[2] = pack2(ar[3].x, ar[3].y); pa1[3] = pack2(ar[3].z, ar[3].w);
        pb0[0] = deq_pair(br[0].x, br[0].y); pb0[1] = deq_pair(br[0].z, br[0].w);
        pb0[2] = deq_pair(br[1].x, br[1].y); pb0[3] = deq_pair(br[1].z, br[1].w);
        pb1[0] = deq_pair(br[2].x, br[2].y); pb1[1] = deq_pair(br[2].z, br[2].w);
        pb1[2] = deq_pair(br[3].x, br[3].y); pb1[3] = deq_pair(br[3].z, br[3].w);
        *(u32x4*)(adst)     = pa0;
        *(u32x4*)(adst + 4) = pa1;
        *(u32x4*)(bdst)     = pb0;
        *(u32x4*)(bdst + 4) = pb1;
    };

    load_regs(0);
    store_lds(0);
    __syncthreads();

    for (int kt = 0; kt < KT; ++kt) {
        const int cur = kt & 1;
        if (kt + 1 < KT) {
            load_regs(kt + 1);
            if (kt + 2 < KT) {
                __builtin_prefetch(xrow + (kt + 2) * BK, 0, 3);
                __builtin_prefetch(irow + (kt + 2) * BK, 0, 3);
            }
        }
        v16h af[2], bf[4];
#pragma unroll
        for (int mi = 0; mi < 2; ++mi)
            af[mi] = frag_a16(Ash[cur], wm * 32 + mi * 16 + l15, hi);
#pragma unroll
        for (int ni = 0; ni < 4; ++ni)
            bf[ni] = frag_b16(Bsh[cur], wn * 64 + ni * 16 + l15, hi);
#pragma unroll
        for (int mi = 0; mi < 2; ++mi)
#pragma unroll
            for (int ni = 0; ni < 4; ++ni)
                acc[mi][ni] = __builtin_amdgcn_wmma_f32_16x16x32_f16(
                    false, af[mi], false, bf[ni],
                    (short)0, acc[mi][ni], false, false);
        if (kt + 1 < KT) store_lds((kt + 1) & 1);
        __syncthreads();
    }

#pragma unroll
    for (int mi = 0; mi < 2; ++mi) {
        const int row0 = bm + wm * 32 + mi * 16 + hi * 8;
#pragma unroll
        for (int ni = 0; ni < 4; ++ni) {
            const int col = bn + wn * 64 + ni * 16 + l15;
            const float bv = bias[col];
            float* op = out + (size_t)row0 * OUT_F + col;
#pragma unroll
            for (int r = 0; r < 8; ++r)
                op[(size_t)r * OUT_F] = acc[mi][ni][r] + bv;
        }
    }
}

extern "C" void kernel_launch(void* const* d_in, const int* in_sizes, int n_in,
                              void* d_out, int out_size, void* d_ws, size_t ws_size,
                              hipStream_t stream) {
    (void)in_sizes; (void)n_in; (void)out_size;
    const float* x      = (const float*)d_in[0];
    const float* levels = (const float*)d_in[1];
    const int*   widx   = (const int*)d_in[2];
    const float* wscale = (const float*)d_in[3];
    const float* bias   = (const float*)d_in[4];
    float* out = (float*)d_out;

    const size_t XH_BYTES = (size_t)M_TOTAL * IN_F * sizeof(_Float16); //  32 MiB
    const size_t WH_BYTES = (size_t)OUT_F   * IN_F * sizeof(_Float16); //  96 MiB
    dim3 grid(M_TOTAL / BM, OUT_F / BN);   // 32 x 96, M fastest for L2 reuse

    if (ws_size >= XH_BYTES + WH_BYTES) {
        _Float16* xh = (_Float16*)d_ws;
        _Float16* wh = (_Float16*)((char*)d_ws + XH_BYTES);
        cvt_x_kernel<<<(M_TOTAL * IN_F) / (8 * 256), 256, 0, stream>>>(x, xh);
        deq_w_kernel<<<OUT_F, 256, 0, stream>>>(widx, levels, wscale, wh);
        q2_gemm_async<<<grid, 256, 0, stream>>>(xh, wh, bias, out);
    } else {
        q2_linear_fused<<<grid, 256, 0, stream>>>(x, levels, widx, wscale, bias, out);
    }
}
